// MoEConnectionProcessor_57200374448217
// MI455X (gfx1250) — matrix operand order repaired and checked by reading, not compiled, not run on hardware
//
#include <hip/hip_runtime.h>

#define B_ROWS 131072
#define S_DIM  256
#define H_DIM  64
#define E_NUM  3
#define CTX_DIM 512
#define CTX_STRIDE 520           // f16 elements; 1040B row stride -> 4-bank skew, conflict-free b128
#define ROWS_PER_WAVE 16
#define WAVES_PER_BLOCK 4
#define ROWS_PER_BLOCK (ROWS_PER_WAVE * WAVES_PER_BLOCK)

typedef __attribute__((ext_vector_type(16))) _Float16 v16h;
typedef __attribute__((ext_vector_type(8)))  _Float16 v8h;
typedef __attribute__((ext_vector_type(4)))  _Float16 v4h;
typedef __attribute__((ext_vector_type(8)))  float    v8f;

static __device__ __forceinline__ v16h cat16(v8h lo, v8h hi) {
    return __builtin_shufflevector(lo, hi, 0,1,2,3,4,5,6,7,8,9,10,11,12,13,14,15);
}

__global__ __launch_bounds__(128)
void moe_gate_kernel(const float* __restrict__ cs,  const float* __restrict__ na,
                     const float* __restrict__ x0,  const float* __restrict__ x1,
                     const float* __restrict__ x2,  const float* __restrict__ gamma,
                     const float* __restrict__ beta,const float* __restrict__ w1,
                     const float* __restrict__ b1,  const float* __restrict__ w2,
                     const float* __restrict__ b2,  float* __restrict__ outc,
                     float* __restrict__ outw)
{
    __shared__ __align__(16) _Float16 w1t[H_DIM * CTX_STRIDE];                      // w1^T as f16 [n][k]
    __shared__ __align__(16) _Float16 ctx[WAVES_PER_BLOCK][ROWS_PER_WAVE * CTX_STRIDE];
    __shared__ float ew[WAVES_PER_BLOCK][ROWS_PER_WAVE][E_NUM];

    const int tid   = threadIdx.x;
    const int wave  = tid >> 5;
    const int lane  = tid & 31;
    const int nbase = lane & 15;      // column / row-in-tile index
    const int mhalf = lane >> 4;      // which 16-lane half
    const int R     = blockIdx.x * ROWS_PER_BLOCK + wave * ROWS_PER_WAVE;

    // ---------------- stage w1 transposed into LDS as f16 ----------------
    for (int idx = tid; idx < CTX_DIM * H_DIM; idx += 128) {
        const int k = idx >> 6;       // 0..511
        const int n = idx & 63;       // 0..63
        w1t[n * CTX_STRIDE + k] = (_Float16)w1[idx];
    }

    // ---------------- LayerNorm + f16 context staging --------------------
    // lane layout: row = nbase, each of the 2 lane-halves covers 128 cols
    const int    row     = nbase;
    const int    colbase = mhalf * 128;
    const size_t goff    = (size_t)(R + row) * S_DIM + colbase;
    const float4* srcA   = (const float4*)(cs + goff);
    const float4* srcN   = (const float4*)(na + goff);

    __builtin_prefetch(x0 + goff, 0, 0);
    __builtin_prefetch(x1 + goff, 0, 0);
    __builtin_prefetch(x2 + goff, 0, 0);

    float s1 = 0.f, s2 = 0.f;
    #pragma unroll 8
    for (int i = 0; i < 32; ++i) {
        const float4 t = srcA[i];
        s1 += t.x + t.y + t.z + t.w;
        s2 += t.x*t.x + t.y*t.y + t.z*t.z + t.w*t.w;
    }
    s1 += __shfl_xor(s1, 16, 32);
    s2 += __shfl_xor(s2, 16, 32);
    const float mean = s1 * (1.0f / 256.0f);
    const float var  = s2 * (1.0f / 256.0f) - mean * mean;
    const float rstd = rsqrtf(var + 1e-5f);

    _Float16* crow = &ctx[wave][row * CTX_STRIDE];
    #pragma unroll 8
    for (int i = 0; i < 32; ++i) {
        const int   c  = colbase + i * 4;
        const float4 t = srcA[i];
        const float4 g = *(const float4*)(gamma + c);
        const float4 bb= *(const float4*)(beta + c);
        v4h h;
        h[0] = (_Float16)((t.x - mean) * rstd * g.x + bb.x);
        h[1] = (_Float16)((t.y - mean) * rstd * g.y + bb.y);
        h[2] = (_Float16)((t.z - mean) * rstd * g.z + bb.z);
        h[3] = (_Float16)((t.w - mean) * rstd * g.w + bb.w);
        *(v4h*)(crow + c) = h;
        const float4 u = srcN[i];
        v4h hn;
        hn[0] = (_Float16)u.x; hn[1] = (_Float16)u.y;
        hn[2] = (_Float16)u.z; hn[3] = (_Float16)u.w;
        *(v4h*)(crow + S_DIM + c) = hn;
    }
    __syncthreads();

    // ---------------- WMMA: [16 x 512] @ [512 x 64] -----------------------
    v8f acc0 = {}, acc1 = {}, acc2 = {}, acc3 = {};
    const _Float16* arow = &ctx[wave][nbase * CTX_STRIDE];   // A: M = lane%16
    const int akoff = 16 * mhalf;                            // A K-slots: {0..15} vs {16..31}

    #pragma unroll 4
    for (int k0 = 0; k0 < CTX_DIM; k0 += 32) {
        const v16h a = cat16(*(const v8h*)(arow + k0 + akoff),
                             *(const v8h*)(arow + k0 + akoff + 8));
        const int kb = k0 + 16 * mhalf;                      // B: contiguous 16 K per lane-half
        const v16h bf0 = cat16(*(const v8h*)(&w1t[(nbase +  0) * CTX_STRIDE + kb]),
                               *(const v8h*)(&w1t[(nbase +  0) * CTX_STRIDE + kb + 8]));
        const v16h bf1 = cat16(*(const v8h*)(&w1t[(nbase + 16) * CTX_STRIDE + kb]),
                               *(const v8h*)(&w1t[(nbase + 16) * CTX_STRIDE + kb + 8]));
        const v16h bf2 = cat16(*(const v8h*)(&w1t[(nbase + 32) * CTX_STRIDE + kb]),
                               *(const v8h*)(&w1t[(nbase + 32) * CTX_STRIDE + kb + 8]));
        const v16h bf3 = cat16(*(const v8h*)(&w1t[(nbase + 48) * CTX_STRIDE + kb]),
                               *(const v8h*)(&w1t[(nbase + 48) * CTX_STRIDE + kb + 8]));
        acc0 = __builtin_amdgcn_wmma_f32_16x16x32_f16(false, a, false, bf0, (short)0, acc0, false, false);
        acc1 = __builtin_amdgcn_wmma_f32_16x16x32_f16(false, a, false, bf1, (short)0, acc1, false, false);
        acc2 = __builtin_amdgcn_wmma_f32_16x16x32_f16(false, a, false, bf2, (short)0, acc2, false, false);
        acc3 = __builtin_amdgcn_wmma_f32_16x16x32_f16(false, a, false, bf3, (short)0, acc3, false, false);
    }

    // ---------------- epilogue: gelu -> [64x3] proj -> softmax ------------
    // C layout: acc[a][j] = h[m = 8*mhalf + j][n = nbase + 16*a]
    float b1v[4], w2v[4][3];
    #pragma unroll
    for (int a = 0; a < 4; ++a) {
        const int n = nbase + 16 * a;
        b1v[a] = b1[n];
        w2v[a][0] = w2[n * 3 + 0];
        w2v[a][1] = w2[n * 3 + 1];
        w2v[a][2] = w2[n * 3 + 2];
    }
    const float b2_0 = b2[0], b2_1 = b2[1], b2_2 = b2[2];

    #pragma unroll
    for (int j = 0; j < 8; ++j) {
        float hv[4] = { acc0[j] + b1v[0], acc1[j] + b1v[1],
                        acc2[j] + b1v[2], acc3[j] + b1v[3] };
        float t0 = 0.f, t1 = 0.f, t2 = 0.f;
        #pragma unroll
        for (int a = 0; a < 4; ++a) {
            const float x = hv[a];
            const float g = 0.5f * x * (1.0f + erff(x * 0.70710678118654752f));
            t0 += g * w2v[a][0];
            t1 += g * w2v[a][1];
            t2 += g * w2v[a][2];
        }
        #pragma unroll
        for (int d = 1; d < 16; d <<= 1) {      // butterfly within each 16-lane half
            t0 += __shfl_xor(t0, d, 32);
            t1 += __shfl_xor(t1, d, 32);
            t2 += __shfl_xor(t2, d, 32);
        }
        t0 += b2_0; t1 += b2_1; t2 += b2_2;
        const float mx = fmaxf(t0, fmaxf(t1, t2));
        const float e0 = __expf(t0 - mx);
        const float e1 = __expf(t1 - mx);
        const float e2 = __expf(t2 - mx);
        const float inv = 1.0f / (e0 + e1 + e2);
        if (nbase == j) {                       // one lane per row per half
            const int m = 8 * mhalf + j;
            ew[wave][m][0] = e0 * inv;
            ew[wave][m][1] = e1 * inv;
            ew[wave][m][2] = e2 * inv;
            float* wo = outw + (size_t)(R + m) * E_NUM;
            wo[0] = e0 * inv; wo[1] = e1 * inv; wo[2] = e2 * inv;
        }
    }
    __syncthreads();

    // ---------------- streaming combine (HBM-bound phase) -----------------
    const float q0 = ew[wave][row][0];
    const float q1 = ew[wave][row][1];
    const float q2 = ew[wave][row][2];
    const float4* p0 = (const float4*)(x0 + goff);
    const float4* p1 = (const float4*)(x1 + goff);
    const float4* p2 = (const float4*)(x2 + goff);
    float4*       po = (float4*)(outc + goff);
    #pragma unroll 8
    for (int i = 0; i < 32; ++i) {
        const float4 a = p0[i], b = p1[i], c = p2[i];
        float4 r;
        r.x = fmaf(q0, a.x, fmaf(q1, b.x, q2 * c.x));
        r.y = fmaf(q0, a.y, fmaf(q1, b.y, q2 * c.y));
        r.z = fmaf(q0, a.z, fmaf(q1, b.z, q2 * c.z));
        r.w = fmaf(q0, a.w, fmaf(q1, b.w, q2 * c.w));
        po[i] = r;
    }
}

extern "C" void kernel_launch(void* const* d_in, const int* in_sizes, int n_in,
                              void* d_out, int out_size, void* d_ws, size_t ws_size,
                              hipStream_t stream) {
    const float* cs    = (const float*)d_in[0];
    const float* na    = (const float*)d_in[1];
    const float* x0    = (const float*)d_in[2];
    const float* x1    = (const float*)d_in[3];
    const float* x2    = (const float*)d_in[4];
    const float* gamma = (const float*)d_in[5];
    const float* beta  = (const float*)d_in[6];
    const float* w1    = (const float*)d_in[7];
    const float* b1    = (const float*)d_in[8];
    const float* w2    = (const float*)d_in[9];
    const float* b2    = (const float*)d_in[10];

    float* outc = (float*)d_out;
    float* outw = outc + (size_t)B_ROWS * S_DIM;

    dim3 grid(B_ROWS / ROWS_PER_BLOCK);   // 2048
    dim3 block(128);                      // 4 waves (wave32)
    moe_gate_kernel<<<grid, block, 0, stream>>>(cs, na, x0, x1, x2, gamma, beta,
                                                w1, b1, w2, b2, outc, outw);
}
